// ASAEPooling_72258529788227
// MI455X (gfx1250) — matrix operand order repaired and checked by reading, not compiled, not run on hardware
//
#include <hip/hip_runtime.h>
#include <hip/hip_bf16.h>
#include <stdint.h>

// ---------------- problem constants (match reference) ----------------
#define NN   50000      // nodes
#define CC   128        // channels
#define ED   16         // edge_dim
#define EE   800000     // real edges (N*DEG)
#define ET   850000     // EE + NN self loops
#define BG   50         // graphs
#define NPG  1000       // nodes per graph
#define KSEL 500        // ceil(0.5*NPG)
#define NOUT 25000      // BG*KSEL
#define MT   (NN / 16)  // 3125 M tiles

// d_out flat offsets (floats), tuple return order
#define OFF_X   0u                       // x_out            [NOUT*CC]
#define OFF_EI  3200000u                 // new_edge_index   [2*ET]
#define OFF_EA  4900000u                 // new_edge_attr    [ET*ED]
#define OFF_BN  18500000u                // batch_new        [NOUT]
#define OFF_PM  18525000u                // perm             [NOUT]
#define OFF_SC  18550000u                // score            [ET]
#define OFF_MK  19400000u                // mask             [ET]

typedef __attribute__((ext_vector_type(16))) __bf16          v16bf;
typedef __attribute__((ext_vector_type(16))) unsigned short  v16u;
typedef __attribute__((ext_vector_type(8)))  float           v8f;

// order-preserving float<->uint encode for atomicMax on floats
__device__ __forceinline__ unsigned fenc(float f) {
    unsigned u = __float_as_uint(f);
    return (u & 0x80000000u) ? ~u : (u | 0x80000000u);
}
__device__ __forceinline__ float fdec(unsigned k) {
    unsigned u = (k & 0x80000000u) ? (k & 0x7FFFFFFFu) : ~k;
    return __uint_as_float(u);
}
// f32 -> bf16 (round to nearest even)
__device__ __forceinline__ unsigned short f2bf(float f) {
    unsigned u = __float_as_uint(f);
    unsigned r = u + 0x7FFFu + ((u >> 16) & 1u);
    return (unsigned short)(r >> 16);
}
__device__ __forceinline__ float wsum(float v) {
#pragma unroll
    for (int o = 16; o > 0; o >>= 1) v += __shfl_xor(v, o, 32);
    return v;
}
__device__ __forceinline__ void edge_sd(int e, const int* ei, int& s, int& d) {
    if (e < EE) { s = ei[e]; d = ei[EE + e]; }
    else        { s = d = e - EE; }
}

// ---------------- kernels ----------------

// init: xq_enc=0 (unsigned-max identity; every (n,c) gets a self-loop update),
// m_enc=0, denom=0, node_map=-1
__global__ void __launch_bounds__(256)
k_init(unsigned* xq_enc, unsigned* m_enc, float* denom, int* node_map) {
    int i = blockIdx.x * blockDim.x + threadIdx.x;
    if (i < NN * CC) xq_enc[i] = 0u;
    if (i < NN) { m_enc[i] = 0u; denom[i] = 0.0f; node_map[i] = -1; }
}

__global__ void __launch_bounds__(256)
k_zero_xsum(float* xsum) {
    int i = blockIdx.x * blockDim.x + threadIdx.x;
    if (i < NN * CC) xsum[i] = 0.0f;
}

// pack lin_W (f32 row-major [k][n]) into bf16 WMMA B-fragment order:
// flat index i = ((wave*4 + kt)*32 + lane)*16 + elem
__global__ void __launch_bounds__(256)
k_convw(const float* __restrict__ w, unsigned short* __restrict__ wpk) {
    int i = blockIdx.x * blockDim.x + threadIdx.x;
    if (i >= CC * CC) return;
    int elem = i & 15;
    int lane = (i >> 4) & 31;
    int frag = i >> 9;               // 0..31
    int wave = frag >> 2, kt = frag & 3;
    int v = elem >> 1, lo = elem & 1;
    int col = wave * 16 + (lane & 15);
    int kb  = kt * 32 + ((v & 4) ? 16 : 0) + (lane >> 4) * 8 + (v & 3) * 2 + lo;
    wpk[i] = f2bf(w[kb * CC + col]);
}

// scatter-max of x[src] into xq_enc[dst] — one wave per edge, 4 ch/lane
__global__ void __launch_bounds__(256)
k_smax(const float* x, const int* ei, unsigned* xq_enc) {
    int gw = (blockIdx.x * blockDim.x + threadIdx.x) >> 5;
    int lane = threadIdx.x & 31;
    if (gw >= ET) return;
    int s, d; edge_sd(gw, ei, s, d);
    const float* xs = x + (size_t)s * CC;
    unsigned* q = xq_enc + (size_t)d * CC;
    __builtin_prefetch(xs, 0, 0);
#pragma unroll
    for (int i = 0; i < 4; ++i) {
        int c = lane + 32 * i;
        atomicMax(&q[c], fenc(xs[c]));
    }
}

// decode xq_enc and pack into bf16 WMMA A-fragment order:
// one thread per (tile, kt, lane) -> 16 contiguous bf16 (32B)
__global__ void __launch_bounds__(256)
k_packA(const unsigned* __restrict__ xq_enc, unsigned short* __restrict__ apk) {
    int t = blockIdx.x * blockDim.x + threadIdx.x;   // tile*128 + kt*32 + lane
    if (t >= MT * 128) return;
    int lane = t & 31;
    int kt   = (t >> 5) & 3;
    int tile = t >> 7;
    int row  = tile * 16 + (lane & 15);
    int khalf = lane >> 4;
    v16u vals;
#pragma unroll
    for (int v = 0; v < 8; ++v) {
        int ka = kt * 32 + ((v & 4) ? 16 : 0) + khalf * 8 + (v & 3) * 2;
        vals[2 * v]     = f2bf(fdec(xq_enc[(size_t)row * CC + ka]));
        vals[2 * v + 1] = f2bf(fdec(xq_enc[(size_t)row * CC + ka + 1]));
    }
    *(v16u*)(apk + (size_t)t * 16) = vals;
}

// GEMM: xq_lin = xq @ lin_W + lin_b  via bf16 WMMA, f32 accum.
// grid.x = MT, block = 256 = 8 waves; wave w -> output cols [16w,16w+16)
__global__ void __launch_bounds__(256)
k_gemm(const unsigned short* __restrict__ apk, const unsigned short* __restrict__ wpk,
       const float* __restrict__ lin_b, float* __restrict__ xq_lin) {
    __shared__ v16u sB[1024];                        // 32KB packed bf16 weights
    int t = threadIdx.x;
    const uint4* gsrc = (const uint4*)wpk;
    uint4* ldst = (uint4*)sB;
#pragma unroll
    for (int i = 0; i < 8; ++i) ldst[t + 256 * i] = gsrc[t + 256 * i];
    __syncthreads();

    int wave = t >> 5, lane = t & 31;
    int tile = blockIdx.x;
    int col  = wave * 16 + (lane & 15);

    float bias = lin_b[col];
    v8f c;
#pragma unroll
    for (int v = 0; v < 8; ++v) c[v] = bias;

    const v16u* ap = (const v16u*)apk + (size_t)tile * 128 + lane;
    const v16u* bp = sB + wave * 128 + lane;
#pragma unroll
    for (int kt = 0; kt < 4; ++kt) {
        v16u ar = ap[kt * 32];       // 2x global_load_b128, coalesced
        v16u br = bp[kt * 32];       // 2x ds_load_b128
        v16bf a = __builtin_bit_cast(v16bf, ar);
        v16bf b = __builtin_bit_cast(v16bf, br);
        c = __builtin_amdgcn_wmma_f32_16x16x32_bf16(false, a, false, b,
                                                    (short)0, c, false, false);
    }
#pragma unroll
    for (int v = 0; v < 8; ++v) {
        int r = tile * 16 + v + ((lane & 16) ? 8 : 0);
        xq_lin[(size_t)r * CC + col] = c[v];
    }
}

// per-node dots: q_dot = xq_lin . attW[0:128], p_dot = x . attW[128:256]
__global__ void __launch_bounds__(256)
k_ndots(const float* xq_lin, const float* x, const float* attW,
        float* q_dot, float* p_dot) {
    int node = (blockIdx.x * blockDim.x + threadIdx.x) >> 5;
    int lane = threadIdx.x & 31;
    if (node >= NN) return;
    float sq = 0.f, sp = 0.f;
#pragma unroll
    for (int i = 0; i < 4; ++i) {
        int c = lane + 32 * i;
        sq += xq_lin[(size_t)node * CC + c] * attW[c];
        sp += x[(size_t)node * CC + c] * attW[CC + c];
    }
    sq = wsum(sq); sp = wsum(sp);
    if (lane == 0) { q_dot[node] = sq; p_dot[node] = sp; }
}

// raw edge score + leaky relu + scatter-max per dst
__global__ void __launch_bounds__(256)
k_escore(const int* ei, const float* ea, const float* attW, const float* att_b,
         const float* q_dot, const float* p_dot, float* raw, unsigned* m_enc) {
    int e = blockIdx.x * blockDim.x + threadIdx.x;
    if (e >= ET) return;
    int s, d; edge_sd(e, ei, s, d);
    float dot = 0.f;
#pragma unroll
    for (int q = 0; q < ED; ++q) {
        float a = (e < EE) ? ea[(size_t)e * ED + q] : 1.0f;
        dot += a * attW[2 * CC + q];
    }
    float sc = q_dot[d] + p_dot[s] + dot + att_b[0];
    sc = (sc > 0.f) ? sc : 0.2f * sc;
    raw[e] = sc;
    atomicMax(&m_enc[d], fenc(sc));
}

// e = exp(s - m[dst]); denom[dst] += e
__global__ void __launch_bounds__(256)
k_eexp(const int* ei, float* esc, const unsigned* m_enc, float* denom) {
    int e = blockIdx.x * blockDim.x + threadIdx.x;
    if (e >= ET) return;
    int s, d; edge_sd(e, ei, s, d); (void)s;
    float ev = __expf(esc[e] - fdec(m_enc[d]));
    esc[e] = ev;
    atomicAdd(&denom[d], ev);
}

// score = e/(denom+eps); x_sum[dst] += x[src]*score — wave per edge
__global__ void __launch_bounds__(256)
k_eagg(const int* ei, const float* x, const float* esc, const float* denom,
       float* x_sum, float* out_score) {
    int gw = (blockIdx.x * blockDim.x + threadIdx.x) >> 5;
    int lane = threadIdx.x & 31;
    if (gw >= ET) return;
    int s, d; edge_sd(gw, ei, s, d);
    float sc = esc[gw] / (denom[d] + 1e-16f);
    if (lane == 0) out_score[gw] = sc;
    const float* xs = x + (size_t)s * CC;
    __builtin_prefetch(xs, 0, 0);
    float* acc = x_sum + (size_t)d * CC;
#pragma unroll
    for (int i = 0; i < 4; ++i) {
        int c = lane + 32 * i;
        atomicAdd(&acc[c], xs[c] * sc);
    }
}

// per-node LEConv terms: a = xs.le1+b1, b = xs.le2, fit_init = xs.le3+b3
__global__ void __launch_bounds__(256)
k_nfit(const float* x_sum, const float* le1W, const float* le1b,
       const float* le2W, const float* le3W, const float* le3b,
       float* a_buf, float* b_buf, float* fit) {
    int node = (blockIdx.x * blockDim.x + threadIdx.x) >> 5;
    int lane = threadIdx.x & 31;
    if (node >= NN) return;
    float s1 = 0.f, s2 = 0.f, s3 = 0.f;
#pragma unroll
    for (int i = 0; i < 4; ++i) {
        int c = lane + 32 * i;
        float v = x_sum[(size_t)node * CC + c];
        s1 += v * le1W[c]; s2 += v * le2W[c]; s3 += v * le3W[c];
    }
    s1 = wsum(s1); s2 = wsum(s2); s3 = wsum(s3);
    if (lane == 0) { a_buf[node] = s1 + le1b[0]; b_buf[node] = s2; fit[node] = s3 + le3b[0]; }
}

// fit[dst] += a[dst] - b[src] + ea.leE_W
__global__ void __launch_bounds__(256)
k_efit(const int* ei, const float* ea, const float* leEW,
       const float* a_buf, const float* b_buf, float* fit) {
    int e = blockIdx.x * blockDim.x + threadIdx.x;
    if (e >= ET) return;
    int s, d; edge_sd(e, ei, s, d);
    float dot = 0.f;
#pragma unroll
    for (int q = 0; q < ED; ++q)
        dot += ((e < EE) ? ea[(size_t)e * ED + q] : 1.0f) * leEW[q];
    atomicAdd(&fit[d], a_buf[d] - b_buf[s] + dot);
}

// fitness = sigmoid(fit); s = tanh(fitness * w/(|w|+eps))
__global__ void __launch_bounds__(256)
k_nsig(const float* fit, const float* sel_w, float* fitness, float* sval) {
    int n = blockIdx.x * blockDim.x + threadIdx.x;
    if (n >= NN) return;
    float f = 1.0f / (1.0f + __expf(-fit[n]));
    fitness[n] = f;
    float w = sel_w[0];
    sval[n] = tanhf(f * w / (fabsf(w) + 1e-16f));
}

// per-graph top-k via 1024-wide bitonic sort (desc by value, asc index tiebreak)
__global__ void __launch_bounds__(512)
k_topk(const float* sval, int* perm, float* out_perm, float* out_batch) {
    __shared__ float sv[1024];
    __shared__ int   si[1024];
    int g = blockIdx.x, t = threadIdx.x;
    for (int i = t; i < 1024; i += 512) {
        if (i < NPG) { sv[i] = sval[g * NPG + i]; si[i] = i; }
        else         { sv[i] = -INFINITY;         si[i] = 0x7FFFFFFF; }
    }
    __syncthreads();
    for (int ksz = 2; ksz <= 1024; ksz <<= 1) {
        for (int j = ksz >> 1; j > 0; j >>= 1) {
            for (int base = 0; base < 1024; base += 512) {
                int i = base + t;
                int ixj = i ^ j;
                if (ixj > i) {
                    float vi = sv[i], vx = sv[ixj];
                    int ii = si[i], ix = si[ixj];
                    bool xGreater = (vx > vi) || (vx == vi && ix < ii);
                    bool iGreater = (vi > vx) || (vi == vx && ii < ix);
                    bool desc = ((i & ksz) == 0);
                    bool sw = desc ? xGreater : iGreater;
                    if (sw) { sv[i] = vx; sv[ixj] = vi; si[i] = ix; si[ixj] = ii; }
                }
            }
            __syncthreads();
        }
    }
    for (int i = t; i < KSEL; i += 512) {
        int p = g * NPG + si[i];
        perm[g * KSEL + i] = p;
        out_perm[g * KSEL + i]  = (float)p;
        out_batch[g * KSEL + i] = (float)g;
    }
}

__global__ void __launch_bounds__(256)
k_postperm(const int* perm, int* node_map) {
    int i = blockIdx.x * blockDim.x + threadIdx.x;
    if (i >= NOUT) return;
    node_map[perm[i]] = i;
}

// x_out[i,c] = x_sum[perm[i],c] * fitness[perm[i]]
__global__ void __launch_bounds__(256)
k_xout(const int* perm, const float* x_sum, const float* fitness, float* out_x) {
    int idx = blockIdx.x * blockDim.x + threadIdx.x;
    if (idx >= NOUT * CC) return;
    int i = idx >> 7, c = idx & 127;
    int p = perm[i];
    out_x[idx] = x_sum[(size_t)p * CC + c] * fitness[p];
}

// FilterEdges + mask output
__global__ void __launch_bounds__(256)
k_filter(const int* ei, const float* ea, const int* node_map, float* out) {
    int e = blockIdx.x * blockDim.x + threadIdx.x;
    if (e >= ET) return;
    int s, d; edge_sd(e, ei, s, d);
    int ns = node_map[s], nd = node_map[d];
    bool mk = (ns >= 0) && (nd >= 0);
    out[OFF_EI + e]      = mk ? (float)ns : -1.0f;
    out[OFF_EI + ET + e] = mk ? (float)nd : -1.0f;
#pragma unroll
    for (int q = 0; q < ED; ++q) {
        float a = (e < EE) ? ea[(size_t)e * ED + q] : 1.0f;
        out[OFF_EA + (size_t)e * ED + q] = mk ? a : 0.0f;
    }
    out[OFF_MK + e] = mk ? 1.0f : 0.0f;
}

// ---------------- host-side launch ----------------
static inline size_t alignup(size_t x) { return (x + 255) & ~size_t(255); }

extern "C" void kernel_launch(void* const* d_in, const int* in_sizes, int n_in,
                              void* d_out, int out_size, void* d_ws, size_t ws_size,
                              hipStream_t stream) {
    (void)in_sizes; (void)n_in; (void)out_size; (void)ws_size;
    const float* x      = (const float*)d_in[0];
    const int*   ei     = (const int*)  d_in[1];
    const float* ea     = (const float*)d_in[2];
    // d_in[3] = batch (implied by fixed layout)
    const float* linW   = (const float*)d_in[4];
    const float* linb   = (const float*)d_in[5];
    const float* attW   = (const float*)d_in[6];
    const float* attb   = (const float*)d_in[7];
    const float* le1W   = (const float*)d_in[8];
    const float* le1b   = (const float*)d_in[9];
    const float* le2W   = (const float*)d_in[10];
    const float* le3W   = (const float*)d_in[11];
    const float* le3b   = (const float*)d_in[12];
    const float* leEW   = (const float*)d_in[13];
    const float* selw   = (const float*)d_in[14];
    float* out = (float*)d_out;

    // workspace carve-up
    char* p = (char*)d_ws;
    size_t off = 0;
    unsigned* xq_enc = (unsigned*)(p + off); off = alignup(off + (size_t)NN * CC * 4);  // reused as x_sum
    float*    xsum   = (float*)xq_enc;
    float*    xq_lin = (float*)(p + off);    off = alignup(off + (size_t)NN * CC * 4);
    unsigned short* wpk = (unsigned short*)(p + off); off = alignup(off + (size_t)CC * CC * 2);
    unsigned short* apk = (unsigned short*)(p + off); off = alignup(off + (size_t)NN * CC * 2);
    float* q_dot  = (float*)(p + off);    off = alignup(off + (size_t)NN * 4);
    float* p_dot  = (float*)(p + off);    off = alignup(off + (size_t)NN * 4);
    unsigned* m_enc = (unsigned*)(p + off); off = alignup(off + (size_t)NN * 4);
    float* denom  = (float*)(p + off);    off = alignup(off + (size_t)NN * 4);
    float* esc    = (float*)(p + off);    off = alignup(off + (size_t)ET * 4);
    float* a_buf  = (float*)(p + off);    off = alignup(off + (size_t)NN * 4);
    float* b_buf  = (float*)(p + off);    off = alignup(off + (size_t)NN * 4);
    float* fit    = (float*)(p + off);    off = alignup(off + (size_t)NN * 4);
    float* fitness= (float*)(p + off);    off = alignup(off + (size_t)NN * 4);
    float* sval   = (float*)(p + off);    off = alignup(off + (size_t)NN * 4);
    int*   perm   = (int*)(p + off);      off = alignup(off + (size_t)NOUT * 4);
    int*   node_map = (int*)(p + off);    off = alignup(off + (size_t)NN * 4);

    const int TB = 256;
    int gNC   = (NN * CC + TB - 1) / TB;        // thread per (node,channel)
    int gEw   = (ET * 32 + TB - 1) / TB;        // wave-per-edge
    int gNw   = (NN * 32 + TB - 1) / TB;        // wave-per-node
    int gE    = (ET + TB - 1) / TB;
    int gN    = (NN + TB - 1) / TB;

    k_init<<<gNC, TB, 0, stream>>>(xq_enc, m_enc, denom, node_map);
    k_convw<<<(CC * CC + TB - 1) / TB, TB, 0, stream>>>(linW, wpk);
    k_smax<<<gEw, TB, 0, stream>>>(x, ei, xq_enc);
    k_packA<<<(MT * 128 + TB - 1) / TB, TB, 0, stream>>>(xq_enc, apk);
    k_gemm<<<MT, TB, 0, stream>>>(apk, wpk, linb, xq_lin);
    k_ndots<<<gNw, TB, 0, stream>>>(xq_lin, x, attW, q_dot, p_dot);
    k_escore<<<gE, TB, 0, stream>>>(ei, ea, attW, attb, q_dot, p_dot, esc, m_enc);
    k_eexp<<<gE, TB, 0, stream>>>(ei, esc, m_enc, denom);
    k_zero_xsum<<<gNC, TB, 0, stream>>>(xsum);                 // xq_enc reused as x_sum
    k_eagg<<<gEw, TB, 0, stream>>>(ei, x, esc, denom, xsum, out + OFF_SC);
    k_nfit<<<gNw, TB, 0, stream>>>(xsum, le1W, le1b, le2W, le3W, le3b, a_buf, b_buf, fit);
    k_efit<<<gE, TB, 0, stream>>>(ei, ea, leEW, a_buf, b_buf, fit);
    k_nsig<<<gN, TB, 0, stream>>>(fit, selw, fitness, sval);
    k_topk<<<BG, 512, 0, stream>>>(sval, perm, out + OFF_PM, out + OFF_BN);
    k_postperm<<<(NOUT + TB - 1) / TB, TB, 0, stream>>>(perm, node_map);
    k_xout<<<(NOUT * CC + TB - 1) / TB, TB, 0, stream>>>(perm, xsum, fitness, out + OFF_X);
    k_filter<<<gE, TB, 0, stream>>>(ei, ea, node_map, out);
}
